// SelfAttention_47364899340447
// MI455X (gfx1250) — matrix operand order repaired and checked by reading, not compiled, not run on hardware
//
#include <hip/hip_runtime.h>
#include <stdint.h>

typedef __attribute__((ext_vector_type(16))) __bf16 v16bf;
typedef __attribute__((ext_vector_type(8)))  float  v8f;
typedef uint32_t __attribute__((may_alias)) u32a;
typedef uint16_t __attribute__((may_alias)) u16a;

constexpr int kB = 4, kN = 4096, kD = 512, kA = 64;

union FragBF { uint32_t u[8]; v16bf v; };

__device__ __forceinline__ uint16_t f2bf(float f) {
  uint32_t u = __builtin_bit_cast(uint32_t, f);
  u += 0x7FFFu + ((u >> 16) & 1u);          // round-to-nearest-even
  return (uint16_t)(u >> 16);
}

// ---------------------------------------------------------------------------
// Stage 1: Q/K/V projections. X[B,N,D]·W[D,A]+b -> bf16 in workspace.
// Q,K stored [B][N][A]; V stored transposed [B][A][N] for stage-2 B-fragments.
// ---------------------------------------------------------------------------
__global__ __launch_bounds__(256)
void qkv_proj_kernel(const float* __restrict__ X,
                     const float* __restrict__ Wq, const float* __restrict__ bq,
                     const float* __restrict__ Wk, const float* __restrict__ bk,
                     const float* __restrict__ Wv, const float* __restrict__ bv,
                     uint16_t* __restrict__ Qws, uint16_t* __restrict__ Kws,
                     uint16_t* __restrict__ Vtws) {
  __shared__ uint16_t Xs[128 * 32];      // [row][k]  bf16, 8 KB
  __shared__ uint16_t Wt[3][64 * 32];    // [a][k]    bf16 (transposed), 12 KB

  const int b     = blockIdx.x >> 5;          // 32 blocks per batch
  const int rbase = (blockIdx.x & 31) * 128;
  const int tid   = threadIdx.x;
  const int lane  = tid & 31;
  const int w     = tid >> 5;
  const int m     = lane & 15;
  const int hi    = (lane >> 4) & 1;
  const int khalfA = hi * 16;                 // A-frag K half
  const int kbaseB = hi * 16;                 // B-frag K half

  const float* Wmtx[3] = {Wq, Wk, Wv};

  v8f acc[3][4];
  const v8f vzero = {0.f, 0.f, 0.f, 0.f, 0.f, 0.f, 0.f, 0.f};
#pragma unroll
  for (int i = 0; i < 3; ++i)
#pragma unroll
    for (int j = 0; j < 4; ++j) acc[i][j] = vzero;

  for (int k0 = 0; k0 < kD; k0 += 32) {
    __syncthreads();                          // protect LDS reuse
    {   // stage X tile: 128 rows x 32 cols, fp32 -> bf16
      int row = tid >> 1;
      int c0  = (tid & 1) * 16;
      const float* src = X + ((size_t)b * kN + rbase + row) * kD + k0 + c0;
      uint16_t* dst = &Xs[row * 32 + c0];
#pragma unroll
      for (int i = 0; i < 16; ++i) dst[i] = f2bf(src[i]);
    }
    {   // stage W slices transposed: Wt[mtx][a][k], 6144 elems / 256 thr
#pragma unroll
      for (int e = 0; e < 24; ++e) {
        int idx = tid * 24 + e;
        int mtx = idx >> 11;
        int rem = idx & 2047;
        int a   = rem >> 5;
        int kk  = rem & 31;
        Wt[mtx][a * 32 + kk] = f2bf(Wmtx[mtx][(size_t)(k0 + kk) * kA + a]);
      }
    }
    __syncthreads();

    // A fragment: 16x32 bf16 (contiguous k = 16*hi + s, same order as B)
    FragBF af;
    const u32a* Xs32 = (const u32a*)Xs;
    const int arow = (w * 16 + m) * 32;
#pragma unroll
    for (int r = 0; r < 4; ++r) af.u[r]     = Xs32[(arow + khalfA + 2 * r) >> 1];
#pragma unroll
    for (int r = 0; r < 4; ++r) af.u[4 + r] = Xs32[(arow + 8 + khalfA + 2 * r) >> 1];

#pragma unroll
    for (int mtx = 0; mtx < 3; ++mtx) {
      const u32a* W32 = (const u32a*)Wt[mtx];
#pragma unroll
      for (int ct = 0; ct < 4; ++ct) {
        FragBF bf;
        const int brow = (ct * 16 + m) * 32;
#pragma unroll
        for (int r = 0; r < 8; ++r) bf.u[r] = W32[(brow + kbaseB + 2 * r) >> 1];
        acc[mtx][ct] = __builtin_amdgcn_wmma_f32_16x16x32_bf16(
            false, af.v, false, bf.v, (short)0, acc[mtx][ct], false, false);
      }
    }
  }

  // epilogue: bias + bf16 stores (C/D layout: row = e + 8*hi, col = lane&15)
#pragma unroll
  for (int ct = 0; ct < 4; ++ct) {
    const int col = ct * 16 + m;
    const float biq = bq[col], bik = bk[col], biv = bv[col];
#pragma unroll
    for (int e = 0; e < 8; ++e) {
      const int row = rbase + w * 16 + e + hi * 8;
      const size_t off = ((size_t)b * kN + row) * kA + col;
      Qws[off] = f2bf(acc[0][ct][e] + biq);
      Kws[off] = f2bf(acc[1][ct][e] + bik);
      Vtws[((size_t)b * kA + col) * kN + row] = f2bf(acc[2][ct][e] + biv);
    }
  }
}

// ---------------------------------------------------------------------------
// Stage 2: flash attention. Each wave: one 16-row query tile; keys streamed
// in 32-key chunks through LDS via global_load_async_to_lds (ASYNCcnt).
// ---------------------------------------------------------------------------
__global__ __launch_bounds__(256)
void attention_kernel(const uint16_t* __restrict__ Qws,
                      const uint16_t* __restrict__ Kws,
                      const uint16_t* __restrict__ Vtws,
                      float* __restrict__ out) {
  __shared__ uint16_t KsT[32 * 64];      // [key][a]   4 KB
  __shared__ uint16_t VsT[64 * 32];      // [a][key]   4 KB
  __shared__ uint16_t Pt[8][16 * 32];    // per-wave P tile, 8 KB

  const int b     = blockIdx.x >> 5;
  const int qbase = (blockIdx.x & 31) * 128;
  const int tid   = threadIdx.x;
  const int lane  = tid & 31;
  const int w     = tid >> 5;
  const int m     = lane & 15;
  const int hi    = (lane >> 4) & 1;
  const int khalfA = hi * 16;
  const int kbaseB = hi * 16;

  // loop-invariant Q fragments (two 16x32 bf16 halves of a-dim)
  FragBF qf[2];
  {
    const u32a* Q32 =
        (const u32a*)(Qws + ((size_t)b * kN + qbase + w * 16 + m) * kA);
#pragma unroll
    for (int h = 0; h < 2; ++h) {
      const int a0 = h * 32;
#pragma unroll
      for (int r = 0; r < 4; ++r) qf[h].u[r]     = Q32[(a0 + khalfA + 2 * r) >> 1];
#pragma unroll
      for (int r = 0; r < 4; ++r) qf[h].u[4 + r] = Q32[(a0 + 8 + khalfA + 2 * r) >> 1];
    }
  }

  v8f o[4];
  const v8f vzero = {0.f, 0.f, 0.f, 0.f, 0.f, 0.f, 0.f, 0.f};
#pragma unroll
  for (int ct = 0; ct < 4; ++ct) o[ct] = vzero;
  float mrow[8], lrow[8];
#pragma unroll
  for (int r = 0; r < 8; ++r) { mrow[r] = -3.0e38f; lrow[r] = 0.f; }

  const uint16_t* Ksrc = Kws  + (size_t)b * kN * kA;
  const uint16_t* Vsrc = Vtws + (size_t)b * kA * kN;
  const uint32_t ks_lds = (uint32_t)(uintptr_t)&KsT[0];
  const uint32_t vs_lds = (uint32_t)(uintptr_t)&VsT[0];

  for (int k0 = 0; k0 < kN; k0 += 32) {
    __syncthreads();                         // all waves done with prev chunk
    {   // K chunk: 32 keys x 64 a = 4096 B contiguous; 16 B per thread
      const uint16_t* src = Ksrc + (size_t)k0 * kA + tid * 8;
      const uint32_t  dst = ks_lds + (uint32_t)tid * 16u;
      asm volatile("global_load_async_to_lds_b128 %0, %1, off"
                   :: "v"(dst), "v"((uint64_t)(uintptr_t)src) : "memory");
    }
    {   // V chunk: 64 a-rows x 32 keys (stride kN); 16 B per thread
      const int row = tid >> 2;
      const int c   = (tid & 3) * 8;
      const uint16_t* src = Vsrc + (size_t)row * kN + k0 + c;
      const uint32_t  dst = vs_lds + (uint32_t)(row * 32 + c) * 2u;
      asm volatile("global_load_async_to_lds_b128 %0, %1, off"
                   :: "v"(dst), "v"((uint64_t)(uintptr_t)src) : "memory");
    }
    asm volatile("s_wait_asynccnt 0" ::: "memory");
    __syncthreads();                         // chunk visible to all waves

    // scores: S = Q · K^T  (contract a-dim, two 16-key subtiles)
    v8f s[2];
    const u32a* K32 = (const u32a*)KsT;
#pragma unroll
    for (int st = 0; st < 2; ++st) {
      v8f c = vzero;
      const int keyrow = (st * 16 + m) * 64;
#pragma unroll
      for (int h = 0; h < 2; ++h) {
        FragBF bf;
#pragma unroll
        for (int r = 0; r < 8; ++r)
          bf.u[r] = K32[(keyrow + h * 32 + kbaseB + 2 * r) >> 1];
        c = __builtin_amdgcn_wmma_f32_16x16x32_bf16(
            false, qf[h].v, false, bf.v, (short)0, c, false, false);
      }
      s[st] = c;
    }

    // streaming softmax (rows live in 16-lane halves of C layout)
    const float scale = 0.125f;              // 1/sqrt(64)
#pragma unroll
    for (int r = 0; r < 8; ++r) {
      float s0 = s[0][r] * scale;
      float s1 = s[1][r] * scale;
      float t = fmaxf(s0, s1);
#pragma unroll
      for (int msk = 1; msk <= 8; msk <<= 1)
        t = fmaxf(t, __shfl_xor(t, msk, 32));
      const float mn = fmaxf(mrow[r], t);
      const float alpha = __expf(mrow[r] - mn);
      const float p0 = __expf(s0 - mn);
      const float p1 = __expf(s1 - mn);
      float rs = p0 + p1;
#pragma unroll
      for (int msk = 1; msk <= 8; msk <<= 1)
        rs += __shfl_xor(rs, msk, 32);
      lrow[r] = lrow[r] * alpha + rs;
      mrow[r] = mn;
      const int prow = r + hi * 8;
      Pt[w][prow * 32 + m]      = f2bf(p0);
      Pt[w][prow * 32 + 16 + m] = f2bf(p1);
#pragma unroll
      for (int ct = 0; ct < 4; ++ct) o[ct][r] *= alpha;
    }

    // P fragment (A layout, 16x32) from this wave's private LDS tile
    FragBF pf;
    const u32a* P32 = (const u32a*)Pt[w];
    const int parow = m * 32;
#pragma unroll
    for (int r = 0; r < 4; ++r) pf.u[r]     = P32[(parow + khalfA + 2 * r) >> 1];
#pragma unroll
    for (int r = 0; r < 4; ++r) pf.u[4 + r] = P32[(parow + 8 + khalfA + 2 * r) >> 1];

    // O += P · V (WMMA C operand carries the rescaled accumulator)
    const u32a* V32 = (const u32a*)VsT;
#pragma unroll
    for (int ct = 0; ct < 4; ++ct) {
      FragBF bf;
      const int vrow = (ct * 16 + m) * 32;
#pragma unroll
      for (int r = 0; r < 8; ++r) bf.u[r] = V32[(vrow + kbaseB + 2 * r) >> 1];
      o[ct] = __builtin_amdgcn_wmma_f32_16x16x32_bf16(
          false, pf.v, false, bf.v, (short)0, o[ct], false, false);
    }
  }

  // epilogue: normalize and store fp32
  float inv[8];
#pragma unroll
  for (int r = 0; r < 8; ++r) inv[r] = 1.0f / lrow[r];
#pragma unroll
  for (int ct = 0; ct < 4; ++ct) {
    const int col = ct * 16 + m;
#pragma unroll
    for (int e = 0; e < 8; ++e) {
      const int row = qbase + w * 16 + e + hi * 8;
      out[((size_t)b * kN + row) * kA + col] = o[ct][e] * inv[e];
    }
  }
}

// ---------------------------------------------------------------------------
extern "C" void kernel_launch(void* const* d_in, const int* in_sizes, int n_in,
                              void* d_out, int out_size, void* d_ws, size_t ws_size,
                              hipStream_t stream) {
  (void)in_sizes; (void)n_in; (void)out_size; (void)ws_size;
  const float* X  = (const float*)d_in[0];
  const float* Wq = (const float*)d_in[1];
  const float* bq = (const float*)d_in[2];
  const float* Wk = (const float*)d_in[3];
  const float* bk = (const float*)d_in[4];
  const float* Wv = (const float*)d_in[5];
  const float* bv = (const float*)d_in[6];
  float* out = (float*)d_out;

  uint16_t* Qws  = (uint16_t*)d_ws;                       // [B][N][A] bf16, 2 MB
  uint16_t* Kws  = Qws + (size_t)kB * kN * kA;            // [B][N][A] bf16, 2 MB
  uint16_t* Vtws = Kws + (size_t)kB * kN * kA;            // [B][A][N] bf16, 2 MB

  const dim3 grid(kB * (kN / 128));
  qkv_proj_kernel<<<grid, 256, 0, stream>>>(X, Wq, bq, Wk, bk, Wv, bv,
                                            Qws, Kws, Vtws);
  attention_kernel<<<grid, 256, 0, stream>>>(Qws, Kws, Vtws, out);
}
